// SMFB_51367808860417
// MI455X (gfx1250) — matrix operand order repaired and checked by reading, not compiled, not run on hardware
//
#include <hip/hip_runtime.h>
#include <hip/hip_bf16.h>

// Problem constants (from the reference)
#define BQ   2
#define DIMQ 128
#define LQ   2304      // H*W = 48*48
#define D2Q  64
#define NSQ  64        // DSTATE
#define KCQ  4
#define CH   16        // scan LDS chunk (l-steps per double-buffer stage)

typedef __attribute__((ext_vector_type(16))) __bf16 v16bf;
typedef __attribute__((ext_vector_type(8)))  float  v8f;
typedef __attribute__((ext_vector_type(8)))  unsigned v8u;
typedef __attribute__((ext_vector_type(4)))  int    v4i;

// Pack two f32 -> two bf16 (RNE) in one dword via v_perm_b32.
__device__ __forceinline__ unsigned pack_bf(float f0, float f1) {
  unsigned u0 = __float_as_uint(f0), u1 = __float_as_uint(f1);
  u0 = u0 + 0x7FFFu + ((u0 >> 16) & 1u);
  u1 = u1 + 0x7FFFu + ((u1 >> 16) & 1u);
  // result = {u1[31:16], u0[31:16]} : bytes 2,3 of S1(u0) then 2,3 of S0(u1)
  return __builtin_amdgcn_perm(u1, u0, 0x07060302);
}

// ---------------------------------------------------------------------------
// Async global->LDS staging (CDNA5 path) with synchronous fallback.
// Builtin prototype (from hipcc diagnostic): (v4i AS1*, v4i AS3*, imm, imm).
// ---------------------------------------------------------------------------
#if __has_builtin(__builtin_amdgcn_global_load_async_to_lds_b128)
#define ASYNC_LDS 1
#endif

__device__ __forceinline__ void copy16_g2l(const float* g, float* l) {
#ifdef ASYNC_LDS
  __builtin_amdgcn_global_load_async_to_lds_b128(
      (__attribute__((address_space(1))) v4i*)(v4i*)g,
      (__attribute__((address_space(3))) v4i*)(v4i*)l, 0, 0);
#else
  *(float4*)l = *(const float4*)g;
#endif
}

__device__ __forceinline__ void wait_async_lds() {
#ifdef ASYNC_LDS
#if __has_builtin(__builtin_amdgcn_s_wait_asynccnt)
  __builtin_amdgcn_s_wait_asynccnt(0);
#else
  asm volatile("s_wait_asynccnt 0" ::: "memory");
#endif
#endif
}

// ---------------------------------------------------------------------------
// 16x16-tile GEMM: O[m,n] = sum_k A[m,k]*B[k,n] + bias[m]
// All strides/dims are compile-time so addressing is 32-bit constant-stride
// and the A loads (ASK==1) merge into global_load_b128.
// CDNA5 bf16 layouts: A row = lane%16, K(e) = kb + (e<8 ? 8*half+e : 8+8*half+e)
//                     B col = lane%16, K(e) = kb + (e<8 ? 8*half+e : 8+8*half+e)
//                     D row = j + 8*half, col = lane%16.
// ---------------------------------------------------------------------------
template <int ASM, int ASK, int BSK, int BSN, int OSM, int OSN, int M, int N, int K>
__global__ void wmma_gemm16(const float* __restrict__ A, const float* __restrict__ Bp,
                            float* __restrict__ O, const float* __restrict__ bias) {
  const int wave = threadIdx.x >> 5;
  const int lane = threadIdx.x & 31;
  constexpr int NT = N >> 4;
  const int tile = blockIdx.x * (blockDim.x >> 5) + wave;
  if (tile >= (M >> 4) * NT) return;          // wave-uniform: EXEC stays all-1s
  const int tm = (tile / NT) << 4;
  const int tn = (tile % NT) << 4;
  const int m = lane & 15;
  const int half = lane >> 4;
  v8f acc = {};
  for (int kb = 0; kb < K; kb += 32) {
    v8u au, bu;
    if constexpr (ASK == 1) {                 // A rows contiguous in K
      const float* arow = A + (unsigned)(tm + m) * ASM + kb + 8 * half;
      const float4 a0 = *(const float4*)(arow);
      const float4 a1 = *(const float4*)(arow + 4);
      const float4 a2 = *(const float4*)(arow + 16);
      const float4 a3 = *(const float4*)(arow + 20);
      au[0] = pack_bf(a0.x, a0.y); au[1] = pack_bf(a0.z, a0.w);
      au[2] = pack_bf(a1.x, a1.y); au[3] = pack_bf(a1.z, a1.w);
      au[4] = pack_bf(a2.x, a2.y); au[5] = pack_bf(a2.z, a2.w);
      au[6] = pack_bf(a3.x, a3.y); au[7] = pack_bf(a3.z, a3.w);
    } else {
#pragma unroll
      for (int e2 = 0; e2 < 8; ++e2) {
        const int e0 = e2 * 2;
        const int ka = kb + ((e0 < 8) ? (8 * half + e0) : (8 + 8 * half + e0));
        au[e2] = pack_bf(A[(unsigned)(tm + m) * ASM + (unsigned)ka * ASK],
                         A[(unsigned)(tm + m) * ASM + (unsigned)(ka + 1) * ASK]);
      }
    }
#pragma unroll
    for (int e2 = 0; e2 < 8; ++e2) {
      const int kr = kb + ((e2 < 4) ? (8 * half + e2 * 2) : (8 + 8 * half + e2 * 2));
      bu[e2] = pack_bf(Bp[(unsigned)kr * BSK + (unsigned)(tn + m) * BSN],
                       Bp[(unsigned)(kr + 1) * BSK + (unsigned)(tn + m) * BSN]);
    }
    acc = __builtin_amdgcn_wmma_f32_16x16x32_bf16(
        false, __builtin_bit_cast(v16bf, au), false, __builtin_bit_cast(v16bf, bu),
        (short)0, acc, false, false);
  }
#pragma unroll
  for (int j = 0; j < 8; ++j) {
    const int row = tm + j + 8 * half;
    const int col = tn + m;
    O[(unsigned)row * OSM + (unsigned)col * OSN] = acc[j] + bias[row];
  }
}

// ---------------------------------------------------------------------------
// Depthwise causal conv (K=4) + SiLU, both directions.
// t layout (b, 128, L); reverse direction reads t flipped along L and stores
// xc already in scan order.  xc layout: (bdir = b*2+dir, d, L).
// ---------------------------------------------------------------------------
__global__ void conv_silu(const float* __restrict__ t,
                          const float* __restrict__ fw, const float* __restrict__ fb,
                          const float* __restrict__ rw, const float* __restrict__ rb,
                          float* __restrict__ xc) {
  const int idx = blockIdx.x * blockDim.x + threadIdx.x;
  if (idx >= 4 * D2Q * LQ) return;
  const int l = idx % LQ;
  const int d = (idx / LQ) & (D2Q - 1);
  const int bdir = idx / (D2Q * LQ);
  const int b = bdir >> 1, dir = bdir & 1;
  const float* w  = dir ? rw : fw;
  const float* cb = dir ? rb : fb;
  const float* urow = t + ((size_t)b * DIMQ + (dir ? D2Q + d : d)) * LQ;
  float s = cb[d];
#pragma unroll
  for (int k = 0; k < KCQ; ++k) {
    const int lp = l - (KCQ - 1) + k;
    if (lp >= 0) {
      const float u = dir ? urow[LQ - 1 - lp] : urow[lp];
      s += w[d * KCQ + k] * u;
    }
  }
  xc[idx] = s / (1.f + __expf(-s));           // SiLU
}

// ---------------------------------------------------------------------------
// Fused Delta/B/C projections for one direction:
//   delta = softplus(xc^T @ Wd + bd),  Bm = xc^T @ WB,  Cm = xc^T @ WC
// A (xc^T tile) is loaded/converted once per k-step, reused by 3 WMMAs.
// Outputs layout (bdir, l, 64).
// ---------------------------------------------------------------------------
__global__ void s6_proj(const float* __restrict__ xc,
                        const float* __restrict__ Wd, const float* __restrict__ bd,
                        const float* __restrict__ WB, const float* __restrict__ WC,
                        float* __restrict__ dOut, float* __restrict__ BOut,
                        float* __restrict__ COut, int dir) {
  const int wave = threadIdx.x >> 5;          // 0..3 -> N tile (64/16)
  const int lane = threadIdx.x & 31;
  const int bdir = blockIdx.y * 2 + dir;
  const int tm = blockIdx.x << 4;             // L tile
  const int tn = wave << 4;                   // feature tile
  const int m = lane & 15, half = lane >> 4;
  const float* xcb = xc + (size_t)bdir * D2Q * LQ;
  v8f aD = {}, aB = {}, aC = {};
  for (int kb = 0; kb < D2Q; kb += 32) {
    v8u au, du, bu, cu;
#pragma unroll
    for (int e2 = 0; e2 < 8; ++e2) {
      const int e0 = e2 * 2;
      const int ka = kb + ((e0 < 8) ? (8 * half + e0) : (8 + 8 * half + e0));
      au[e2] = pack_bf(xcb[(unsigned)ka * LQ + (tm + m)],
                       xcb[(unsigned)(ka + 1) * LQ + (tm + m)]);
      const int kr = ka;
      du[e2] = pack_bf(Wd[(unsigned)kr * NSQ + (tn + m)], Wd[(unsigned)(kr + 1) * NSQ + (tn + m)]);
      bu[e2] = pack_bf(WB[(unsigned)kr * NSQ + (tn + m)], WB[(unsigned)(kr + 1) * NSQ + (tn + m)]);
      cu[e2] = pack_bf(WC[(unsigned)kr * NSQ + (tn + m)], WC[(unsigned)(kr + 1) * NSQ + (tn + m)]);
    }
    const v16bf a = __builtin_bit_cast(v16bf, au);
    aD = __builtin_amdgcn_wmma_f32_16x16x32_bf16(false, a, false, __builtin_bit_cast(v16bf, du), (short)0, aD, false, false);
    aB = __builtin_amdgcn_wmma_f32_16x16x32_bf16(false, a, false, __builtin_bit_cast(v16bf, bu), (short)0, aB, false, false);
    aC = __builtin_amdgcn_wmma_f32_16x16x32_bf16(false, a, false, __builtin_bit_cast(v16bf, cu), (short)0, aC, false, false);
  }
#pragma unroll
  for (int j = 0; j < 8; ++j) {
    const int l = tm + j + 8 * half;
    const int f = tn + m;
    const size_t o = ((size_t)bdir * LQ + l) * NSQ + f;
    const float dv = aD[j] + bd[f];
    dOut[o] = dv > 20.f ? dv : log1pf(__expf(dv));     // softplus
    BOut[o] = aB[j];
    COut[o] = aC[j];
  }
}

// ---------------------------------------------------------------------------
// Selective scan: h[l] = exp(delta*A)*h[l-1] + (delta*x)*B[l];
// y[l,d] = sum_n h*C[l,n] + D[d]*x.
// One wave per (bdir, d); lanes hold n = {2*lane, 2*lane+1}.
// Bm/Cm/Delta rows are shared by all 8 waves of the block -> double-buffered
// into LDS in CH-step chunks with async global->LDS copies (ASYNCcnt).
// Reverse direction writes its output flipped back into place (c = 64+d).
// ---------------------------------------------------------------------------
__global__ void s6_scan(const float* __restrict__ xc, const float* __restrict__ dl,
                        const float* __restrict__ Bm, const float* __restrict__ Cm,
                        const float* __restrict__ fAl, const float* __restrict__ fDp,
                        const float* __restrict__ rAl, const float* __restrict__ rDp,
                        float* __restrict__ y) {
  __shared__ float Bs[2][CH][NSQ];
  __shared__ float Cs[2][CH][NSQ];
  __shared__ float Ds[2][CH][NSQ];
  const int tid = threadIdx.x;
  const int wave = tid >> 5, lane = tid & 31;
  const int bdir = blockIdx.x >> 3;                 // 4 (b,dir) combos
  const int d = ((blockIdx.x & 7) << 3) + wave;     // 64 d rows
  const int b = bdir >> 1, dir = bdir & 1;
  const float* Al = dir ? rAl : fAl;
  const float  Dd = (dir ? rDp : fDp)[d];
  const float A1 = -__expf(Al[d * NSQ + 2 * lane]);
  const float A2 = -__expf(Al[d * NSQ + 2 * lane + 1]);
  const float* xrow = xc + ((size_t)bdir * D2Q + d) * LQ;
  const float* Bg = Bm + (size_t)bdir * LQ * NSQ;
  const float* Cg = Cm + (size_t)bdir * LQ * NSQ;
  const float* Dg = dl + (size_t)bdir * LQ * NSQ;
  const int c = dir ? D2Q + d : d;
  float* yrow = y + ((size_t)b * DIMQ + c) * LQ;

  // Each of the 256 threads stages one 16B segment per matrix per chunk.
  const int crow = tid >> 4;                        // 0..15 (l within chunk)
  const int ccol = (tid & 15) << 2;                 // 0,4,...,60 (n)
  auto stage = [&](int l0, int buf) {
    const size_t go = (size_t)(l0 + crow) * NSQ + ccol;
    copy16_g2l(Bg + go, &Bs[buf][crow][ccol]);
    copy16_g2l(Cg + go, &Cs[buf][crow][ccol]);
    copy16_g2l(Dg + go, &Ds[buf][crow][ccol]);
  };

  stage(0, 0);
  wait_async_lds();
  __syncthreads();

  float h1 = 0.f, h2 = 0.f;
  for (int ck = 0; ck < LQ / CH; ++ck) {
    const int buf = ck & 1;
    if (ck + 1 < LQ / CH) stage((ck + 1) * CH, buf ^ 1);   // prefetch next chunk
    const int l0 = ck * CH;
    for (int li = 0; li < CH; ++li) {
      const int l = l0 + li;
      const float  de = Ds[buf][li][d & (NSQ - 1)];
      const float  xv = xrow[l];
      const float2 Bv = *(const float2*)&Bs[buf][li][2 * lane];
      const float2 Cv = *(const float2*)&Cs[buf][li][2 * lane];
      const float  dx = de * xv;
      h1 = __expf(de * A1) * h1 + dx * Bv.x;
      h2 = __expf(de * A2) * h2 + dx * Bv.y;
      float p = h1 * Cv.x + h2 * Cv.y;
#pragma unroll
      for (int off = 16; off > 0; off >>= 1) p += __shfl_xor(p, off, 32);
      if (lane == 0) yrow[dir ? (LQ - 1 - l) : l] = p + Dd * xv;
    }
    wait_async_lds();
    __syncthreads();
  }
}

// ---------------------------------------------------------------------------
extern "C" void kernel_launch(void* const* d_in, const int* in_sizes, int n_in,
                              void* d_out, int out_size, void* d_ws, size_t ws_size,
                              hipStream_t stream) {
  const float* x   = (const float*)d_in[0];
  const float* Wx  = (const float*)d_in[1];
  const float* bx  = (const float*)d_in[2];
  const float* Wp  = (const float*)d_in[3];
  const float* bp  = (const float*)d_in[4];
  const float* fcw = (const float*)d_in[5];
  const float* fcb = (const float*)d_in[6];
  const float* fWd = (const float*)d_in[7];
  const float* fbd = (const float*)d_in[8];
  const float* fWB = (const float*)d_in[9];
  const float* fWC = (const float*)d_in[10];
  const float* fAl = (const float*)d_in[11];
  const float* fDp = (const float*)d_in[12];
  const float* rcw = (const float*)d_in[13];
  const float* rcb = (const float*)d_in[14];
  const float* rWd = (const float*)d_in[15];
  const float* rbd = (const float*)d_in[16];
  const float* rWB = (const float*)d_in[17];
  const float* rWC = (const float*)d_in[18];
  const float* rAl = (const float*)d_in[19];
  const float* rDp = (const float*)d_in[20];
  float* out = (float*)d_out;

  // Workspace carve-up (floats). Total = 5 * 589824 * 4B ~= 11.8 MB.
  float* ws    = (float*)d_ws;
  const size_t SEQ = (size_t)DIMQ * LQ;           // 294912 per batch
  float* t     = ws;                              // (B,128,L)  in-proj result
  float* xc    = t  + BQ * SEQ;                   // (4,64,L)   conv+SiLU, scan order
  float* delta = xc + 4 * (size_t)D2Q * LQ;       // (4,L,64)
  float* Bmat  = delta + 4 * (size_t)LQ * NSQ;    // (4,L,64)
  float* Cmat  = Bmat  + 4 * (size_t)LQ * NSQ;    // (4,L,64)
  float* y     = t;                               // reuse: t dead after conv

  // 1) Input projection: t[b] = Wx(128x128) @ x[b](128xL) + bx
  for (int b = 0; b < BQ; ++b)
    wmma_gemm16<DIMQ, 1, LQ, 1, LQ, 1, DIMQ, LQ, DIMQ>
        <<<144, 256, 0, stream>>>(Wx, x + b * SEQ, t + b * SEQ, bx);

  // 2) Depthwise conv + SiLU (both directions; reverse pre-flipped)
  conv_silu<<<(4 * D2Q * LQ + 255) / 256, 256, 0, stream>>>(t, fcw, fcb, rcw, rcb, xc);

  // 3) Fused Delta/B/C projections per direction
  s6_proj<<<dim3(LQ / 16, BQ), 128, 0, stream>>>(xc, fWd, fbd, fWB, fWC,
                                                 delta, Bmat, Cmat, 0);
  s6_proj<<<dim3(LQ / 16, BQ), 128, 0, stream>>>(xc, rWd, rbd, rWB, rWC,
                                                 delta, Bmat, Cmat, 1);

  // 4) Selective scan (async-LDS double buffering; writes y in concat layout)
  s6_scan<<<32, 256, 0, stream>>>(xc, delta, Bmat, Cmat, fAl, fDp, rAl, rDp, y);

  // 5) Output projection: out[b] = Wp @ y[b] + bp
  for (int b = 0; b < BQ; ++b)
    wmma_gemm16<DIMQ, 1, LQ, 1, LQ, 1, DIMQ, LQ, DIMQ>
        <<<144, 256, 0, stream>>>(Wp, y + b * SEQ, out + b * SEQ, bp);
}
